// BcosAttention_71562745085987
// MI455X (gfx1250) — matrix operand order repaired and checked by reading, not compiled, not run on hardware
//
#include <hip/hip_runtime.h>
#include <hip/hip_bf16.h>

// ---------------------------------------------------------------------------
// Types for CDNA5 WMMA (wave32): V_WMMA_F32_16X16X32_BF16
// ---------------------------------------------------------------------------
typedef __attribute__((ext_vector_type(16))) __bf16 v16bf;
typedef __attribute__((ext_vector_type(8)))  __bf16 v8bf;
typedef __attribute__((ext_vector_type(8)))  float  v8f;

__device__ __forceinline__ v8f wmma_bf16(v16bf a, v16bf b, v8f c) {
  // (neg_a, A, neg_b, B, c_mod, C, reuse_a, reuse_b)
  return __builtin_amdgcn_wmma_f32_16x16x32_bf16(false, a, false, b,
                                                 (short)0, c, false, false);
}

__device__ __forceinline__ v16bf combine16(v8bf lo, v8bf hi) {
  v16bf r;
#pragma unroll
  for (int i = 0; i < 8; ++i) { r[i] = lo[i]; r[i + 8] = hi[i]; }
  return r;
}

// A-fragment (16x32 MxK, bf16): lane l holds row l%16.
// half = (lane>=16): 16 contiguous K elements at {16*half..}.
__device__ __forceinline__ v16bf load_a16(const __bf16* rowp, int half) {
  v8bf lo = *(const v8bf*)(rowp + 16 * half);
  v8bf hi = *(const v8bf*)(rowp + 16 * half + 8);
  return combine16(lo, hi);
}

// B-fragment (32x16 KxN, bf16): lane l holds 16 contiguous K-elements of
// row l%16 of the (N,K) row-major operand, at K offset 16*half.
__device__ __forceinline__ v16bf load_b16(const __bf16* p) {
  v8bf lo = *(const v8bf*)(p);
  v8bf hi = *(const v8bf*)(p + 8);
  return combine16(lo, hi);
}

// CDNA5 async global->LDS (ASYNCcnt path). LDS address = low 32 bits of the
// generic shared pointer (flat aperture truncation rule).
__device__ __forceinline__ void async_load_b128(void* lds_ptr, const void* gptr) {
  unsigned lds = (unsigned)(uintptr_t)lds_ptr;
  unsigned long long ga = (unsigned long long)(uintptr_t)gptr;
  asm volatile("global_load_async_to_lds_b128 %0, %1, off"
               :: "v"(lds), "v"(ga) : "memory");
}
__device__ __forceinline__ void wait_async0() {
  asm volatile("s_wait_asynccnt 0x0" ::: "memory");
}

#define SQRT512F 22.62741699796952f

// ---------------------------------------------------------------------------
// Kernel 1: per-row L2 norm + bf16 conversion.  One wave per 512-wide row.
// mode 0: out = row / ||row||  (weight pre-normalization)
// mode 1: out = row, norms[row] = ||row||  (activations)
// ---------------------------------------------------------------------------
__global__ __launch_bounds__(256)
void rownorm_bf16_kernel(const float* __restrict__ in, __bf16* __restrict__ out,
                         float* __restrict__ norms, int rows, int mode) {
  const int wave = threadIdx.x >> 5;
  const int lane = threadIdx.x & 31;
  const int row = blockIdx.x * 8 + wave;
  if (row >= rows) return;
  const float* p = in + (size_t)row * 512;
  float vals[16];
  float ss = 0.f;
#pragma unroll
  for (int i = 0; i < 16; ++i) {
    float v = p[lane + 32 * i];
    vals[i] = v;
    ss += v * v;
  }
#pragma unroll
  for (int m = 16; m >= 1; m >>= 1) ss += __shfl_xor(ss, m, 32);
  const float nrm = sqrtf(ss);
  const float scale = (mode == 0) ? (1.0f / fmaxf(nrm, 1e-20f)) : 1.0f;
  __bf16* o = out + (size_t)row * 512;
#pragma unroll
  for (int i = 0; i < 16; ++i) o[lane + 32 * i] = (__bf16)(vals[i] * scale);
  if (mode == 1 && lane == 0) norms[row] = nrm;
}

// ---------------------------------------------------------------------------
// Kernel 2: bf16 GEMM  C(MxN,f32) = A(MxK,bf16) * B(NxK,bf16)^T
// Block tile 64x128, BK=64, 256 threads = 8 waves (2x4), wave tile 32x32.
// Tiles staged via global_load_async_to_lds_b128.  N,K compile-time.
// ---------------------------------------------------------------------------
#define G_BM 64
#define G_BN 128
#define G_BK 64
#define G_LDS 72   // padded bf16 stride (144B: 16B aligned, bank-spread)

template <int N, int K>
__global__ __launch_bounds__(256)
void gemm_bf16_kernel(const __bf16* __restrict__ A, const __bf16* __restrict__ B,
                      float* __restrict__ C) {
  __shared__ __bf16 sA[G_BM * G_LDS];
  __shared__ __bf16 sB[G_BN * G_LDS];

  const int tid  = threadIdx.x;
  const int lane = tid & 31;
  const int wave = tid >> 5;
  const int wm = wave >> 2;        // 0..1
  const int wn = wave & 3;         // 0..3
  const int half = lane >> 4;
  const int l16  = lane & 15;
  const int bm0 = blockIdx.y * G_BM;
  const int bn0 = blockIdx.x * G_BN;

  // Precompute per-thread async-copy coordinates (16B chunks of 8 bf16).
  const __bf16* agp[2]; void* alds[2];
#pragma unroll
  for (int t = 0; t < 2; ++t) {            // A: 64 rows x 8 chunks = 512
    int idx = tid + t * 256;
    int r = idx >> 3, ch = idx & 7;
    agp[t]  = A + (size_t)(bm0 + r) * K + ch * 8;
    alds[t] = &sA[r * G_LDS + ch * 8];
  }
  const __bf16* bgp[4]; void* blds[4];
#pragma unroll
  for (int t = 0; t < 4; ++t) {            // B: 128 rows x 8 chunks = 1024
    int idx = tid + t * 256;
    int r = idx >> 3, ch = idx & 7;
    bgp[t]  = B + (size_t)(bn0 + r) * K + ch * 8;
    blds[t] = &sB[r * G_LDS + ch * 8];
  }

  v8f acc[2][2] = {};

  for (int k0 = 0; k0 < K; k0 += G_BK) {
#pragma unroll
    for (int t = 0; t < 2; ++t) async_load_b128(alds[t], agp[t] + k0);
#pragma unroll
    for (int t = 0; t < 4; ++t) async_load_b128(blds[t], bgp[t] + k0);
    wait_async0();
    __syncthreads();

#pragma unroll
    for (int ks = 0; ks < 2; ++ks) {
      v16bf af[2], bfr[2];
#pragma unroll
      for (int i = 0; i < 2; ++i)
        af[i] = load_a16(&sA[(wm * 32 + i * 16 + l16) * G_LDS + ks * 32], half);
#pragma unroll
      for (int j = 0; j < 2; ++j)
        bfr[j] = load_b16(&sB[(wn * 32 + j * 16 + l16) * G_LDS + ks * 32 + half * 16]);
#pragma unroll
      for (int i = 0; i < 2; ++i)
#pragma unroll
        for (int j = 0; j < 2; ++j)
          acc[i][j] = wmma_bf16(af[i], bfr[j], acc[i][j]);
    }
    __syncthreads();   // compute done before next iteration's async writes
  }

  // C layout: reg r, lane l -> row r + 8*half, col l%16.  Compile-time N
  // makes r*N an immediate store offset.
#pragma unroll
  for (int i = 0; i < 2; ++i)
#pragma unroll
    for (int j = 0; j < 2; ++j) {
      float* cp = C + (size_t)(bm0 + wm * 32 + i * 16 + 8 * half) * N +
                  bn0 + wn * 32 + j * 16 + l16;
#pragma unroll
      for (int r = 0; r < 8; ++r) cp[r * N] = acc[i][j][r];
    }
}

// ---------------------------------------------------------------------------
// Kernel 3: QKV bcos epilogue: maxout over (j, j+1536), o|o|/(||x|| sqrt512).
// q,k scattered as (B*H, N, Dh); v scattered TRANSPOSED as (B*H, Dh, N) so
// the attention kernel can async-copy V^T tiles without an LDS transpose.
// ---------------------------------------------------------------------------
__global__ __launch_bounds__(256)
void qkv_epilogue_kernel(const float* __restrict__ pre, const float* __restrict__ xnorm,
                         __bf16* __restrict__ q, __bf16* __restrict__ k,
                         __bf16* __restrict__ vT) {
  const int idx = blockIdx.x * 256 + threadIdx.x;
  if (idx >= 4096 * 1536) return;
  const int row = idx / 1536, j = idx - row * 1536;
  const float a = pre[(size_t)row * 3072 + j];
  const float b = pre[(size_t)row * 3072 + j + 1536];
  const float m = fmaxf(a, b);
  const float o = m * fabsf(m) / (xnorm[row] * SQRT512F);
  const int s = j / 512, c = j - s * 512;
  const int h = c >> 6, d = c & 63;
  const int bb = row >> 11, n = row & 2047;
  const int bh = bb * 8 + h;
  const __bf16 ov = (__bf16)o;
  if (s == 0)      q[((size_t)bh * 2048 + n) * 64 + d] = ov;
  else if (s == 1) k[((size_t)bh * 2048 + n) * 64 + d] = ov;
  else             vT[((size_t)bh * 64 + d) * 2048 + n] = ov;
}

// ---------------------------------------------------------------------------
// Kernel 4: flash attention (bf16 WMMA, f32 online softmax).
// grid = (N/64, B*H), 128 threads = 4 waves; wave owns 16 query rows.
// K and V^T tiles staged with async global->LDS copies.
// ---------------------------------------------------------------------------
__global__ __launch_bounds__(128)
void flash_attn_kernel(const __bf16* __restrict__ Q, const __bf16* __restrict__ Kmat,
                       const __bf16* __restrict__ Vt, float* __restrict__ Out, int N) {
  __shared__ __bf16 sK[32 * 72];        // keys x Dh (padded)
  __shared__ __bf16 sVt[64 * 40];       // Dh x keys (padded)
  __shared__ __bf16 sP[4][16 * 40];     // per-wave P tile 16x32

  const int tid = threadIdx.x, lane = tid & 31, wave = tid >> 5;
  const int half = lane >> 4, l16 = lane & 15;
  const int bh = blockIdx.y;
  const int q0 = blockIdx.x * 64 + wave * 16;
  const size_t base  = (size_t)bh * N * 64;   // (bh, n, d) for Q/K
  const size_t basev = (size_t)bh * 64 * N;   // (bh, d, n) for V^T

  // Per-thread async-copy coords.  K tile: 32 rows x 8 chunks = 256.
  const __bf16* kg[2]; void* klds[2];
  const __bf16* vg[2]; void* vlds[2];
#pragma unroll
  for (int t = 0; t < 2; ++t) {
    int idx = tid + t * 128;
    int r = idx >> 3, ch = idx & 7;            // K: row 0..31, chunk 0..7
    kg[t]   = Kmat + base + (size_t)r * 64 + ch * 8;
    klds[t] = &sK[r * 72 + ch * 8];
    int vr = idx >> 2, vch = idx & 3;          // V^T: row 0..63, chunk 0..3
    vg[t]   = Vt + basev + (size_t)vr * N + vch * 8;
    vlds[t] = &sVt[vr * 40 + vch * 8];
  }

  // Q A-fragments for Dh=64 (two K-steps of 32)
  v16bf aq[2];
  {
    const __bf16* qrow = Q + base + (size_t)(q0 + l16) * 64;
#pragma unroll
    for (int ks = 0; ks < 2; ++ks) aq[ks] = load_a16(qrow + ks * 32, half);
  }

  v8f acc_o[4] = {};
  float m_i[8], l_i[8];
#pragma unroll
  for (int r = 0; r < 8; ++r) { m_i[r] = -1e30f; l_i[r] = 0.f; }
  const float scl = 0.125f;   // 1/sqrt(64)

  for (int kb = 0; kb < N; kb += 32) {
#pragma unroll
    for (int t = 0; t < 2; ++t) async_load_b128(klds[t], kg[t] + (size_t)kb * 64);
#pragma unroll
    for (int t = 0; t < 2; ++t) async_load_b128(vlds[t], vg[t] + kb);
    wait_async0();
    __syncthreads();

    // S = Q x K^T  (16 x 32), C-fragment layout
    v8f s[2] = {};
#pragma unroll
    for (int j = 0; j < 2; ++j)
#pragma unroll
      for (int ks = 0; ks < 2; ++ks) {
        v16bf bk = load_b16(&sK[(j * 16 + l16) * 72 + ks * 32 + half * 16]);
        s[j] = wmma_bf16(aq[ks], bk, s[j]);
      }

    // online softmax; xor masks 1..8 stay within each 16-lane half
    float pr[2][8];
#pragma unroll
    for (int r = 0; r < 8; ++r) {
      float s0 = s[0][r] * scl, s1 = s[1][r] * scl;
      float mx = fmaxf(s0, s1);
#pragma unroll
      for (int m = 8; m >= 1; m >>= 1) mx = fmaxf(mx, __shfl_xor(mx, m, 32));
      const float newm = fmaxf(m_i[r], mx);
      const float corr = __expf(m_i[r] - newm);
      const float p0 = __expf(s0 - newm), p1 = __expf(s1 - newm);
      float rs = p0 + p1;
#pragma unroll
      for (int m = 8; m >= 1; m >>= 1) rs += __shfl_xor(rs, m, 32);
      l_i[r] = l_i[r] * corr + rs;
      m_i[r] = newm;
#pragma unroll
      for (int t = 0; t < 4; ++t) acc_o[t][r] *= corr;
      pr[0][r] = p0; pr[1][r] = p1;
    }

    // C-layout P -> per-wave LDS -> A-fragment (same-wave DS is in-order)
#pragma unroll
    for (int j = 0; j < 2; ++j)
#pragma unroll
      for (int r = 0; r < 8; ++r)
        sP[wave][(r + 8 * half) * 40 + j * 16 + l16] = (__bf16)pr[j][r];

    v16bf ap = load_a16(&sP[wave][l16 * 40], half);
#pragma unroll
    for (int t = 0; t < 4; ++t) {
      v16bf bv = load_b16(&sVt[(t * 16 + l16) * 40 + half * 16]);
      acc_o[t] = wmma_bf16(ap, bv, acc_o[t]);
    }
    __syncthreads();
  }

  // epilogue: divide by row sum, store to (B,N,C) f32 with immediate offsets
  const int b = bh >> 3, h = bh & 7;
  float* op = Out + ((size_t)(b * 2048 + q0 + 8 * half)) * 512 + h * 64 + l16;
#pragma unroll
  for (int t = 0; t < 4; ++t)
#pragma unroll
    for (int r = 0; r < 8; ++r)
      op[r * 512 + t * 16] = acc_o[t][r] / l_i[r];
}

// ---------------------------------------------------------------------------
// Kernel 5: projection bcos epilogue -> final f32 output (B,N,C)
// ---------------------------------------------------------------------------
__global__ __launch_bounds__(256)
void proj_epilogue_kernel(const float* __restrict__ pre, const float* __restrict__ anorm,
                          float* __restrict__ out) {
  const int idx = blockIdx.x * 256 + threadIdx.x;
  if (idx >= 4096 * 512) return;
  const int row = idx >> 9, j = idx & 511;
  const float a = pre[(size_t)row * 1024 + j];
  const float b = pre[(size_t)row * 1024 + j + 512];
  const float m = fmaxf(a, b);
  out[idx] = m * fabsf(m) / (anorm[row] * SQRT512F);
}

// ---------------------------------------------------------------------------
// Host launcher
// ---------------------------------------------------------------------------
extern "C" void kernel_launch(void* const* d_in, const int* in_sizes, int n_in,
                              void* d_out, int out_size, void* d_ws, size_t ws_size,
                              hipStream_t stream) {
  const float* x     = (const float*)d_in[0];   // (2,2048,512)
  const float* Wqkv  = (const float*)d_in[1];   // (3072,512)
  const float* Wproj = (const float*)d_in[2];   // (1024,512)

  char* ws = (char*)d_ws;
  size_t off = 0;
  auto take = [&](size_t bytes) -> void* {
    void* p = ws + off;
    off = (off + bytes + 255) & ~(size_t)255;
    return p;
  };

  __bf16* wqkv_bf  = (__bf16*)take((size_t)3072 * 512 * 2);
  __bf16* wproj_bf = (__bf16*)take((size_t)1024 * 512 * 2);
  __bf16* x_bf     = (__bf16*)take((size_t)4096 * 512 * 2);
  float*  x_norm   = (float*)take((size_t)4096 * 4);
  float*  a_norm   = (float*)take((size_t)4096 * 4);
  __bf16* qb       = (__bf16*)take((size_t)16 * 2048 * 64 * 2);
  __bf16* kb       = (__bf16*)take((size_t)16 * 2048 * 64 * 2);
  __bf16* vTb      = (__bf16*)take((size_t)16 * 64 * 2048 * 2);
  float*  attn_f   = (float*)take((size_t)4096 * 512 * 4);
  __bf16* attn_bf  = (__bf16*)take((size_t)4096 * 512 * 2);
  float*  pre      = (float*)take((size_t)4096 * 3072 * 4);  // reused for proj pre

  // 1) weight normalization + bf16; x bf16 + row norms
  rownorm_bf16_kernel<<<3072 / 8, 256, 0, stream>>>(Wqkv, wqkv_bf, nullptr, 3072, 0);
  rownorm_bf16_kernel<<<1024 / 8, 256, 0, stream>>>(Wproj, wproj_bf, nullptr, 1024, 0);
  rownorm_bf16_kernel<<<4096 / 8, 256, 0, stream>>>(x, x_bf, x_norm, 4096, 1);

  // 2) QKV GEMM: (4096x512) x (3072x512)^T -> (4096x3072)
  gemm_bf16_kernel<3072, 512><<<dim3(3072 / G_BN, 4096 / G_BM), 256, 0, stream>>>(
      x_bf, wqkv_bf, pre);

  // 3) bcos maxout + scatter to q/k/vT
  qkv_epilogue_kernel<<<(4096 * 1536 + 255) / 256, 256, 0, stream>>>(
      pre, x_norm, qb, kb, vTb);

  // 4) flash attention per (b,h)
  flash_attn_kernel<<<dim3(2048 / 64, 16), 128, 0, stream>>>(
      qb, kb, vTb, attn_f, 2048);

  // 5) attention-output norms + bf16
  rownorm_bf16_kernel<<<4096 / 8, 256, 0, stream>>>(attn_f, attn_bf, a_norm, 4096, 1);

  // 6) projection GEMM: (4096x512) x (1024x512)^T -> (4096x1024)
  gemm_bf16_kernel<1024, 512><<<dim3(1024 / G_BN, 4096 / G_BM), 256, 0, stream>>>(
      attn_bf, wproj_bf, pre);

  // 7) final bcos epilogue
  proj_epilogue_kernel<<<(4096 * 512 + 255) / 256, 256, 0, stream>>>(
      pre, a_norm, (float*)d_out);
}
